// HyperbolicAttention_21431886807644
// MI455X (gfx1250) — matrix operand order repaired and checked by reading, not compiled, not run on hardware
//
#include <hip/hip_runtime.h>
#include <hip/hip_bf16.h>
#include <math.h>

#define EPS_F      1e-5f
#define MAX_NORM_F (1.0f - 1e-3f)

#define B_DIM  2
#define S_DIM  512
#define D_DIM  512
#define H_DIM  8
#define DH_DIM 64
#define NTOK   (B_DIM * S_DIM)   // 1024

typedef __attribute__((ext_vector_type(16))) _Float16 v16h;
typedef __attribute__((ext_vector_type(8)))  _Float16 v8h;
typedef __attribute__((ext_vector_type(8)))  float    v8f;
typedef __attribute__((ext_vector_type(4)))  unsigned int u32x4;
typedef __attribute__((ext_vector_type(8)))  unsigned int u32x8;

// ---------------------------------------------------------------------------
// Tensor Data Mover: async DMA of a 2-D f16 tile (rows x dim0, row stride
// stride0 elements) from global memory into LDS (row-major, contiguous).
// D# layout per CDNA5 ISA. Groups 2/3 zero (2-D).
// Caller must be a single wave; follow with s_wait_tensorcnt + barrier.
// ---------------------------------------------------------------------------
__device__ __forceinline__ void tdm_load_2d_f16(const void* gsrc, void* ldst,
                                                unsigned dim0, unsigned rows,
                                                unsigned stride0) {
  const unsigned long long ga = (unsigned long long)gsrc;
  const unsigned lds = (unsigned)(unsigned long long)ldst;  // low 32 bits give the LDS byte address
  u32x4 g0;
  g0[0] = 1u;                                   // count=1 (valid), no gather
  g0[1] = lds;                                  // lds_addr (bytes)
  g0[2] = (unsigned)ga;                         // global_addr[31:0]
  g0[3] = (unsigned)((ga >> 32) & 0x01FFFFFFull) | (2u << 30);  // addr[56:32]|type=2
  u32x8 g1;
  g1[0] = 1u << 16;                             // data_size=1 (2 bytes)
  g1[1] = (dim0 & 0xFFFFu) << 16;               // tensor_dim0[15:0]
  g1[2] = (dim0 >> 16) | ((rows & 0xFFFFu) << 16);  // dim0 hi | tensor_dim1 lo
  g1[3] = (rows >> 16) | ((dim0 & 0xFFFFu) << 16);  // dim1 hi | tile_dim0
  g1[4] = rows & 0xFFFFu;                       // tile_dim1 (tile_dim2=0)
  g1[5] = stride0;                              // tensor_dim0_stride[31:0]
  g1[6] = 0u;
  g1[7] = 0u;
  u32x4 gz;
  gz[0] = 0u; gz[1] = 0u; gz[2] = 0u; gz[3] = 0u;
  asm volatile("tensor_load_to_lds %0, %1, %2, %3"
               :
               : "s"(g0), "s"(g1), "s"(gz), "s"(gz)
               : "memory");
}

// ---------------------------------------------------------------------------
// WMMA fragment helpers (CDNA5 wave32 layouts, CDNA5 ISA §7.12.2)
// ---------------------------------------------------------------------------
// A: 16x32 f16, lane = M (lane&15); element j holds K = (j>>3)*16 + (lane>>4)*8 + (j&7)
__device__ __forceinline__ v16h load_a_frag(const _Float16* __restrict__ base,
                                            int lda, int lane) {
  const int m  = lane & 15;
  const int kb = (lane >> 4) << 3;  // 0 or 8
  const _Float16* p = base + (size_t)m * lda + kb;
  v8h lo = *(const v8h*)p;          // K = kb..kb+7
  v8h hi = *(const v8h*)(p + 16);   // K = 16+kb..16+kb+7
  v16h a;
#pragma unroll
  for (int i = 0; i < 8; ++i) { a[i] = lo[i]; a[i + 8] = hi[i]; }
  return a;
}

// B: 32x16 f16, lane = N (lane&15); element j holds K = (j>>3)*16 + (lane>>4)*8 + (j&7).
// Source: row-major B-column data (W[n, k]) -- works for global or LDS ptrs.
__device__ __forceinline__ v16h load_b_frag(const _Float16* base, int ldb,
                                            int lane) {
  const int n  = lane & 15;
  const int kb = (lane >> 4) << 3;  // 0 or 8
  const _Float16* p = base + (size_t)n * ldb + kb;
  v8h lo = *(const v8h*)p;
  v8h hi = *(const v8h*)(p + 16);
  v16h b;
#pragma unroll
  for (int i = 0; i < 8; ++i) { b[i] = lo[i]; b[i + 8] = hi[i]; }
  return b;
}

// ---------------------------------------------------------------------------
// Kernel 1: logmap0 over full rows (D=512): out = atanh(min(n,1-eps)) * x / n
// ---------------------------------------------------------------------------
__global__ __launch_bounds__(256) void logmap_rows_kernel(
    const float* __restrict__ in, _Float16* __restrict__ out) {
  __shared__ float red[256];
  const int row = blockIdx.x;
  const int t   = threadIdx.x;
  const float* src = in + (size_t)row * D_DIM;
  const float v0 = src[t];
  const float v1 = src[t + 256];
  red[t] = v0 * v0 + v1 * v1;
  __syncthreads();
  for (int off = 128; off > 0; off >>= 1) {
    if (t < off) red[t] += red[t + off];
    __syncthreads();
  }
  const float nrm   = fmaxf(sqrtf(red[0]), EPS_F);
  const float scale = atanhf(fminf(nrm, 1.0f - EPS_F)) / nrm;
  _Float16* dst = out + (size_t)row * D_DIM;
  dst[t]       = (_Float16)(v0 * scale);
  dst[t + 256] = (_Float16)(v1 * scale);
}

// ---------------------------------------------------------------------------
// Kernel 2: f32 -> f16 convert (weights)
// ---------------------------------------------------------------------------
__global__ void f32_to_f16_kernel(const float* __restrict__ in,
                                  _Float16* __restrict__ out, int n) {
  const int i = blockIdx.x * blockDim.x + threadIdx.x;
  if (i < n) out[i] = (_Float16)in[i];
}

// ---------------------------------------------------------------------------
// Kernel 3: WMMA GEMM  C[m,n] = sum_k A[m,k]*W[n,k] + bias[n]
// Block = 256 threads = 8 waves; wave computes a 16x64 strip; the shared
// 64xK B strip is TDM-staged into LDS once per block (removes the 8x
// redundant global B traffic; WMMA feeds from ds_load).
// Grid: (N/64, M/128). Requires K == 512 (LDS tile sizing).
// ---------------------------------------------------------------------------
__global__ __launch_bounds__(256) void wmma_gemm_bias_kernel(
    const _Float16* __restrict__ A, const _Float16* __restrict__ W,
    const float* __restrict__ bias, float* __restrict__ outF32,
    _Float16* __restrict__ outF16, int M, int N, int K) {
  __shared__ _Float16 btile[64 * 512];  // 64 KB
  const int lane = threadIdx.x & 31;
  const int wave = threadIdx.x >> 5;
  const int m0 = blockIdx.y * 128 + wave * 16;
  const int n0 = blockIdx.x * 64;

  if (wave == 0) {
    tdm_load_2d_f16(W + (size_t)n0 * K, btile, (unsigned)K, 64u, (unsigned)K);
    __builtin_amdgcn_s_wait_tensorcnt(0);
  }
  __syncthreads();

  v8f acc[4] = {v8f{}, v8f{}, v8f{}, v8f{}};
  for (int k0 = 0; k0 < K; k0 += 32) {
    __builtin_prefetch(A + (size_t)m0 * K + k0 + 64, 0, 3);
    v16h a = load_a_frag(A + (size_t)m0 * K + k0, K, lane);
#pragma unroll
    for (int t = 0; t < 4; ++t) {
      v16h b = load_b_frag(btile + (size_t)(t * 16) * K + k0, K, lane);
      acc[t] = __builtin_amdgcn_wmma_f32_16x16x32_f16(
          false, a, false, b, (short)0, acc[t], false, false);
    }
  }

  const int nl = lane & 15;
  const int mo = (lane >> 4) << 3;
#pragma unroll
  for (int t = 0; t < 4; ++t) {
    const int nn = n0 + t * 16 + nl;
    const float bv = bias ? bias[nn] : 0.0f;
#pragma unroll
    for (int r = 0; r < 8; ++r) {
      const float val = acc[t][r] + bv;
      const size_t idx = (size_t)(m0 + mo + r) * N + nn;
      if (outF32) outF32[idx] = val;
      if (outF16) outF16[idx] = (_Float16)val;
    }
  }
}

// ---------------------------------------------------------------------------
// Kernel 4: per-head expmap0 (dh=64 segments) + store ||y||^2 per (b,h,s)
// ---------------------------------------------------------------------------
__global__ __launch_bounds__(512) void expmap_heads_kernel(
    const float* __restrict__ in, _Float16* __restrict__ outh,
    float* __restrict__ norm2) {
  __shared__ float red[512];
  const int row = blockIdx.x;          // 0..NTOK-1
  const int b = row / S_DIM;
  const int s = row % S_DIM;
  const int t = threadIdx.x;
  const int h = t >> 6;
  const float v = in[(size_t)row * D_DIM + t];
  red[t] = v * v;
  __syncthreads();
  for (int off = 32; off > 0; off >>= 1) {
    if ((t & 63) < off) red[t] += red[t + off];
    __syncthreads();
  }
  const float nv = sqrtf(red[h << 6]);
  const float n  = fmaxf(nv, EPS_F);
  float sc = tanhf(n) / n;
  float ny = sc * nv;
  if (ny > MAX_NORM_F) { sc *= MAX_NORM_F / ny; ny = MAX_NORM_F; }
  outh[(size_t)row * D_DIM + t] = (_Float16)(v * sc);
  if ((t & 63) == 0)
    norm2[(size_t)(b * H_DIM + h) * S_DIM + s] = ny * ny;
}

// ---------------------------------------------------------------------------
// Kernel 5: transpose V per head to f16: Vt[b,h,d,s] = vp[b,s,h*64+d]
// ---------------------------------------------------------------------------
__global__ void transpose_v_kernel(const float* __restrict__ vp,
                                   _Float16* __restrict__ vt) {
  const int idx = blockIdx.x * blockDim.x + threadIdx.x;
  if (idx >= B_DIM * H_DIM * DH_DIM * S_DIM) return;
  const int s    = idx & (S_DIM - 1);
  const int rest = idx >> 9;
  const int d    = rest & (DH_DIM - 1);
  const int bh   = rest >> 6;
  const int b    = bh >> 3;
  const int h    = bh & 7;
  vt[idx] = (_Float16)vp[(size_t)(b * S_DIM + s) * D_DIM + h * DH_DIM + d];
}

// ---------------------------------------------------------------------------
// Kernel 6: fused Q·K^T (WMMA) -> hyperbolic distance -> softmax -> P (f16)
// Grid: (S/16, H, B); block = 256 threads = 8 waves; 16x512 score tile in LDS.
// ---------------------------------------------------------------------------
__global__ __launch_bounds__(256) void scores_softmax_kernel(
    const _Float16* __restrict__ qh, const _Float16* __restrict__ kh,
    const float* __restrict__ qn2, const float* __restrict__ kn2,
    _Float16* __restrict__ P) {
  __shared__ float smem[16 * S_DIM];
  __shared__ float part[16][16];
  const int lane = threadIdx.x & 31;
  const int wave = threadIdx.x >> 5;
  const int b = blockIdx.z, h = blockIdx.y;
  const int m0 = blockIdx.x * 16;
  const size_t qbase = (size_t)(b * S_DIM + m0) * D_DIM + h * DH_DIM;

  v8f acc[4] = {v8f{}, v8f{}, v8f{}, v8f{}};
#pragma unroll
  for (int k0 = 0; k0 < DH_DIM; k0 += 32) {
    v16h a = load_a_frag(qh + qbase + k0, D_DIM, lane);
#pragma unroll
    for (int t = 0; t < 4; ++t) {
      const int n0 = (wave * 4 + t) * 16;
      v16h bm = load_b_frag(
          kh + (size_t)(b * S_DIM + n0) * D_DIM + h * DH_DIM + k0, D_DIM, lane);
      acc[t] = __builtin_amdgcn_wmma_f32_16x16x32_f16(
          false, a, false, bm, (short)0, acc[t], false, false);
    }
  }

  // dot -> hyperbolic distance -> score, into LDS
  const int nl = lane & 15;
  const int mo = (lane >> 4) << 3;
  const float* qn = qn2 + (size_t)(b * H_DIM + h) * S_DIM;
  const float* kn = kn2 + (size_t)(b * H_DIM + h) * S_DIM;
#pragma unroll
  for (int t = 0; t < 4; ++t) {
    const int ncol = (wave * 4 + t) * 16 + nl;
    const float y2 = kn[ncol];
#pragma unroll
    for (int r = 0; r < 8; ++r) {
      const int mloc = mo + r;
      const float x2  = qn[m0 + mloc];
      const float dot = acc[t][r];
      // mobius_add(-x, y): xy' = <-x,y> = -dot
      const float xy    = -dot;
      const float alpha = 1.0f + 2.0f * xy + y2;
      const float beta  = 1.0f - x2;
      const float num2  = alpha * alpha * x2 + beta * beta * y2
                        + 2.0f * alpha * beta * xy;
      const float den = fmaxf(1.0f + 2.0f * xy + x2 * y2, EPS_F);
      float nr = sqrtf(fmaxf(num2, 0.0f)) / den;
      nr = fminf(nr, MAX_NORM_F);                    // _project
      nr = fminf(fmaxf(nr, EPS_F), 1.0f - EPS_F);    // clip
      const float dist = 2.0f * atanhf(nr);
      smem[mloc * S_DIM + ncol] = -dist * 0.125f;    // /sqrt(dh)
    }
  }
  __syncthreads();

  // softmax: 16 threads per row, 32 columns each
  const int row = threadIdx.x >> 4;
  const int sub = threadIdx.x & 15;
  float mx = -1e30f;
  for (int c = sub * 32; c < sub * 32 + 32; ++c)
    mx = fmaxf(mx, smem[row * S_DIM + c]);
  part[row][sub] = mx;
  __syncthreads();
  if (sub == 0) {
    float m = -1e30f;
    for (int i = 0; i < 16; ++i) m = fmaxf(m, part[row][i]);
    part[row][0] = m;
  }
  __syncthreads();
  const float rmax = part[row][0];
  float sum = 0.0f;
  for (int c = sub * 32; c < sub * 32 + 32; ++c) {
    const float e = __expf(smem[row * S_DIM + c] - rmax);
    smem[row * S_DIM + c] = e;
    sum += e;
  }
  __syncthreads();
  part[row][sub] = sum;
  __syncthreads();
  if (sub == 0) {
    float tt = 0.0f;
    for (int i = 0; i < 16; ++i) tt += part[row][i];
    part[row][0] = tt;
  }
  __syncthreads();
  const float inv = 1.0f / part[row][0];
  _Float16* prow =
      P + (size_t)((b * H_DIM + h) * S_DIM + m0 + row) * S_DIM;
  for (int c = sub * 32; c < sub * 32 + 32; ++c)
    prow[c] = (_Float16)(smem[row * S_DIM + c] * inv);
}

// ---------------------------------------------------------------------------
// Kernel 7: attn @ V per head (WMMA); the shared 64x512 V^T strip is
// TDM-staged into LDS once per block. Grid: (4, H, B); wave = 16x64 tile.
// ---------------------------------------------------------------------------
__global__ __launch_bounds__(256) void wmma_av_kernel(
    const _Float16* __restrict__ P, const _Float16* __restrict__ Vt,
    _Float16* __restrict__ out16) {
  __shared__ _Float16 vtile[64 * 512];  // 64 KB
  const int lane = threadIdx.x & 31;
  const int wave = threadIdx.x >> 5;
  const int b = blockIdx.z, h = blockIdx.y;
  const int bh = b * H_DIM + h;
  const int m0 = (blockIdx.x * 8 + wave) * 16;
  const _Float16* Pb = P + (size_t)bh * S_DIM * S_DIM;

  if (wave == 0) {
    tdm_load_2d_f16(Vt + (size_t)bh * DH_DIM * S_DIM, vtile,
                    (unsigned)S_DIM, (unsigned)DH_DIM, (unsigned)S_DIM);
    __builtin_amdgcn_s_wait_tensorcnt(0);
  }
  __syncthreads();

  v8f acc[4] = {v8f{}, v8f{}, v8f{}, v8f{}};
  for (int k0 = 0; k0 < S_DIM; k0 += 32) {
    __builtin_prefetch(Pb + (size_t)m0 * S_DIM + k0 + 64, 0, 3);
    v16h a = load_a_frag(Pb + (size_t)m0 * S_DIM + k0, S_DIM, lane);
#pragma unroll
    for (int t = 0; t < 4; ++t) {
      v16h bm = load_b_frag(vtile + (size_t)(t * 16) * S_DIM + k0, S_DIM, lane);
      acc[t] = __builtin_amdgcn_wmma_f32_16x16x32_f16(
          false, a, false, bm, (short)0, acc[t], false, false);
    }
  }

  const int nl = lane & 15;
  const int mo = (lane >> 4) << 3;
#pragma unroll
  for (int t = 0; t < 4; ++t) {
#pragma unroll
    for (int r = 0; r < 8; ++r) {
      out16[(size_t)(b * S_DIM + m0 + mo + r) * D_DIM + h * DH_DIM +
            t * 16 + nl] = (_Float16)acc[t][r];
    }
  }
}

// ---------------------------------------------------------------------------
// Kernel 8: final expmap0 over full rows (D=512), f32 out
// ---------------------------------------------------------------------------
__global__ __launch_bounds__(256) void expmap_full_kernel(
    const float* __restrict__ in, float* __restrict__ out) {
  __shared__ float red[256];
  const int row = blockIdx.x;
  const int t   = threadIdx.x;
  const float* src = in + (size_t)row * D_DIM;
  const float v0 = src[t];
  const float v1 = src[t + 256];
  red[t] = v0 * v0 + v1 * v1;
  __syncthreads();
  for (int off = 128; off > 0; off >>= 1) {
    if (t < off) red[t] += red[t + off];
    __syncthreads();
  }
  const float nv = sqrtf(red[0]);
  const float n  = fmaxf(nv, EPS_F);
  float sc = tanhf(n) / n;
  const float ny = sc * nv;
  if (ny > MAX_NORM_F) sc *= MAX_NORM_F / ny;
  float* dst = out + (size_t)row * D_DIM;
  dst[t]       = v0 * sc;
  dst[t + 256] = v1 * sc;
}

// ---------------------------------------------------------------------------
extern "C" void kernel_launch(void* const* d_in, const int* in_sizes, int n_in,
                              void* d_out, int out_size, void* d_ws,
                              size_t ws_size, hipStream_t stream) {
  (void)in_sizes; (void)n_in; (void)out_size; (void)ws_size;
  const float* q  = (const float*)d_in[0];
  const float* k  = (const float*)d_in[1];
  const float* v  = (const float*)d_in[2];
  const float* Wq = (const float*)d_in[3];
  const float* bq = (const float*)d_in[4];
  const float* Wk = (const float*)d_in[5];
  const float* bk = (const float*)d_in[6];
  const float* Wv = (const float*)d_in[7];
  const float* bv = (const float*)d_in[8];
  const float* Wo = (const float*)d_in[9];
  const float* bo = (const float*)d_in[10];
  float* out = (float*)d_out;

  char* ws = (char*)d_ws;
  size_t off = 0;
  auto alloc = [&](size_t bytes) -> void* {
    void* p = ws + off;
    off = (off + bytes + 255) & ~(size_t)255;
    return p;
  };

  _Float16* qt16 = (_Float16*)alloc((size_t)NTOK * D_DIM * 2);
  _Float16* kt16 = (_Float16*)alloc((size_t)NTOK * D_DIM * 2);
  _Float16* vt16 = (_Float16*)alloc((size_t)NTOK * D_DIM * 2);
  _Float16* Wq16 = (_Float16*)alloc((size_t)D_DIM * D_DIM * 2);
  _Float16* Wk16 = (_Float16*)alloc((size_t)D_DIM * D_DIM * 2);
  _Float16* Wv16 = (_Float16*)alloc((size_t)D_DIM * D_DIM * 2);
  _Float16* Wo16 = (_Float16*)alloc((size_t)D_DIM * D_DIM * 2);
  float*    qp   = (float*)alloc((size_t)NTOK * D_DIM * 4);
  float*    kp   = (float*)alloc((size_t)NTOK * D_DIM * 4);
  float*    vp   = (float*)alloc((size_t)NTOK * D_DIM * 4);
  _Float16* qh16 = (_Float16*)alloc((size_t)NTOK * D_DIM * 2);
  _Float16* kh16 = (_Float16*)alloc((size_t)NTOK * D_DIM * 2);
  float*    qn2  = (float*)alloc((size_t)B_DIM * H_DIM * S_DIM * 4);
  float*    kn2  = (float*)alloc((size_t)B_DIM * H_DIM * S_DIM * 4);
  _Float16* vtt  = (_Float16*)alloc((size_t)B_DIM * H_DIM * DH_DIM * S_DIM * 2);
  _Float16* Pm   = (_Float16*)alloc((size_t)B_DIM * H_DIM * S_DIM * S_DIM * 2);
  _Float16* ao16 = (_Float16*)alloc((size_t)NTOK * D_DIM * 2);
  float*    op32 = (float*)alloc((size_t)NTOK * D_DIM * 4);

  // 1) logmap0 into f16 tangent vectors
  logmap_rows_kernel<<<NTOK, 256, 0, stream>>>(q, qt16);
  logmap_rows_kernel<<<NTOK, 256, 0, stream>>>(k, kt16);
  logmap_rows_kernel<<<NTOK, 256, 0, stream>>>(v, vt16);

  // 2) weights -> f16
  const int nW = D_DIM * D_DIM;
  f32_to_f16_kernel<<<(nW + 255) / 256, 256, 0, stream>>>(Wq, Wq16, nW);
  f32_to_f16_kernel<<<(nW + 255) / 256, 256, 0, stream>>>(Wk, Wk16, nW);
  f32_to_f16_kernel<<<(nW + 255) / 256, 256, 0, stream>>>(Wv, Wv16, nW);
  f32_to_f16_kernel<<<(nW + 255) / 256, 256, 0, stream>>>(Wo, Wo16, nW);

  // 3) projections (WMMA + TDM-staged B): x @ W^T + b
  dim3 gg(D_DIM / 64, NTOK / 128);
  wmma_gemm_bias_kernel<<<gg, 256, 0, stream>>>(qt16, Wq16, bq, qp, nullptr,
                                                NTOK, D_DIM, D_DIM);
  wmma_gemm_bias_kernel<<<gg, 256, 0, stream>>>(kt16, Wk16, bk, kp, nullptr,
                                                NTOK, D_DIM, D_DIM);
  wmma_gemm_bias_kernel<<<gg, 256, 0, stream>>>(vt16, Wv16, bv, vp, nullptr,
                                                NTOK, D_DIM, D_DIM);

  // 4) per-head expmap0 for q,k (+ squared norms); transpose v to f16
  expmap_heads_kernel<<<NTOK, 512, 0, stream>>>(qp, qh16, qn2);
  expmap_heads_kernel<<<NTOK, 512, 0, stream>>>(kp, kh16, kn2);
  const int nV = B_DIM * H_DIM * DH_DIM * S_DIM;
  transpose_v_kernel<<<(nV + 255) / 256, 256, 0, stream>>>(vp, vtt);

  // 5) fused QK^T (WMMA) + hyperbolic distance + softmax
  scores_softmax_kernel<<<dim3(S_DIM / 16, H_DIM, B_DIM), 256, 0, stream>>>(
      qh16, kh16, qn2, kn2, Pm);

  // 6) attn @ V (WMMA + TDM-staged V^T)
  wmma_av_kernel<<<dim3(4, H_DIM, B_DIM), 256, 0, stream>>>(Pm, vtt, ao16);

  // 7) output projection (WMMA + TDM-staged B) + final expmap0
  wmma_gemm_bias_kernel<<<gg, 256, 0, stream>>>(ao16, Wo16, bo, op32, nullptr,
                                                NTOK, D_DIM, D_DIM);
  expmap_full_kernel<<<NTOK, 256, 0, stream>>>(op32, out);
}
